// Attention_13030930776649
// MI455X (gfx1250) — matrix operand order repaired and checked by reading, not compiled, not run on hardware
//
#include <hip/hip_runtime.h>
#include <hip/hip_bf16.h>

// ---------------- problem dims (compile-time) ----------------
#define B_DIM 32
#define L_DIM 2048
#define D_DIM 1024
#define MTILE 128          // rows of enc_proj per workgroup (8 waves x 16)
#define NCHUNK 64          // columns per N-chunk (4 WMMA n-tiles)
#define KBLOCK 128         // K block staged in LDS for W1 (double buffered)

typedef __attribute__((ext_vector_type(16))) __bf16 v16bf;
typedef __attribute__((ext_vector_type(8)))  float  v8f;
typedef unsigned int u32x4 __attribute__((ext_vector_type(4)));
typedef unsigned int u32x8 __attribute__((ext_vector_type(8)));

// LDS layout (dynamic shared):
//   sA   : 128 rows x 1024 bf16       = 262144 B
//   sB   : 2 x (64 rows x 128 bf16)   =  32768 B  (double buffer, 16 KB each)
//   sDPB : 1024 f32                   =   4096 B
//   sV   : 1024 f32                   =   4096 B
//   sS   :  128 f32                   =    512 B
#define SMEM_A     0
#define SMEM_B     262144
#define SMEM_BBUF  16384
#define SMEM_DPB   (262144 + 32768)
#define SMEM_V     (SMEM_DPB + 4096)
#define SMEM_S     (SMEM_V + 4096)
#define SMEM_BYTES (SMEM_S + 512)

// stable fast tanh: 1 - 2/(e^{2x}+1); NaN-free, saturates to +/-1
__device__ __forceinline__ float fast_tanh(float x) {
    float e = __expf(2.0f * x);
    return 1.0f - 2.0f / (e + 1.0f);
}

// ---------------- K1: W1 fp32 -> bf16 (same [e][d] layout) ----------------
__global__ __launch_bounds__(256) void cvt_w1_kernel(const float* __restrict__ W1,
                                                     __bf16* __restrict__ w1bf) {
    int idx = blockIdx.x * 256 + threadIdx.x;      // 262144 threads, 4 elems each
    float4 f = ((const float4*)W1)[idx];
    union { __bf16 h[4]; uint2 u; } cv;
    cv.h[0] = (__bf16)f.x; cv.h[1] = (__bf16)f.y;
    cv.h[2] = (__bf16)f.z; cv.h[3] = (__bf16)f.w;
    ((uint2*)w1bf)[idx] = cv.u;
}

// ---------------- K2: dpb[b][e] = h_dec[b]@W2[e] + b1[e] + b2[e] ----------
__global__ __launch_bounds__(256) void decproj_kernel(const float* __restrict__ h_dec,
                                                      const float* __restrict__ W2,
                                                      const float* __restrict__ b1,
                                                      const float* __restrict__ b2,
                                                      float* __restrict__ dpb) {
    int idx = blockIdx.x * 256 + threadIdx.x;      // 32768 outputs
    int b = idx >> 10, e = idx & 1023;
    const float4* hd = (const float4*)(h_dec + (size_t)b * D_DIM);
    const float4* wr = (const float4*)(W2 + (size_t)e * D_DIM);
    float acc = 0.0f;
#pragma unroll 4
    for (int d = 0; d < D_DIM / 4; ++d) {
        float4 a = hd[d], c = wr[d];
        acc += a.x * c.x + a.y * c.y + a.z * c.z + a.w * c.w;
    }
    dpb[idx] = acc + b1[e] + b2[e];
}

// --------- TDM: one descriptor moves a 64x128 bf16 W1 tile into LDS -------
// D# group1 (constant across all blocks): 2D tensor, data_size=2B,
// tensor_dim0=1024, tensor_dim1=1024, tile_dim0=128, tile_dim1=64,
// tensor_dim0_stride=1024, no padding / iterate / multicast.
__device__ __forceinline__ u32x8 tdm_group1() {
    u32x8 g1;
    g1[0] = 0x00010000u;       // workgroup_mask=0, data_size=1 (2 bytes)
    g1[1] = 0x04000000u;       // tensor_dim0 = 1024 (bits 79:48, low half)
    g1[2] = 0x04000000u;       // tensor_dim1 = 1024 (bits 111:80, low half)
    g1[3] = 128u << 16;        // tile_dim0 = 128   (bits 127:112)
    g1[4] = 64u;               // tile_dim1 = 64    (bits 143:128); tile_dim2=0
    g1[5] = 1024u;             // tensor_dim0_stride = 1024 (bits 207:160)
    g1[6] = 0u;
    g1[7] = 0u;
    return g1;
}

__device__ __forceinline__ void tdm_load_b_block(const char* src, unsigned ldsDst,
                                                 u32x8 g1) {
    unsigned long long ga = (unsigned long long)(uintptr_t)src;
    u32x4 g0;
    g0.x = 1u;                                         // count=1, user descriptor
    g0.y = ldsDst;                                     // LDS byte address
    g0.z = (unsigned)ga;                               // global_addr[31:0]
    g0.w = (unsigned)(ga >> 32) | 0x80000000u;         // global_addr[56:32], type=2
    asm volatile("tensor_load_to_lds %0, %1" :: "s"(g0), "s"(g1) : "memory");
}

// ---------------- K3: WMMA scores kernel ---------------------------------
// scores[b][l] = sum_e tanh( (h_enc[b,l]@W1^T)[e] + dpb[b][e] ) * V[e]
__global__ __launch_bounds__(256) void attn_scores_kernel(const float* __restrict__ h_enc,
                                                          const __bf16* __restrict__ w1bf,
                                                          const float* __restrict__ dpb,
                                                          const float* __restrict__ Vvec,
                                                          float* __restrict__ scores) {
    extern __shared__ char smem[];
    char*  sA   = smem + SMEM_A;
    char*  sB   = smem + SMEM_B;
    float* sDPB = (float*)(smem + SMEM_DPB);
    float* sV   = (float*)(smem + SMEM_V);
    float* sS   = (float*)(smem + SMEM_S);
    const unsigned ldsB = __builtin_amdgcn_groupstaticsize() + SMEM_B;  // abs LDS offset

    const int t    = threadIdx.x;
    const int b    = blockIdx.y;
    const int l0   = blockIdx.x * MTILE;
    const int wv   = t >> 5;          // wave id 0..7, owns rows [16wv,16wv+16)
    const int lane = t & 31;
    const int n    = lane & 15;
    const int g    = lane >> 4;

    // ---- stage dpb row, V, zero score accumulators ----
    for (int i = t; i < D_DIM; i += 256) {
        sDPB[i] = dpb[(size_t)b * D_DIM + i];
        sV[i]   = Vvec[i];
    }
    if (t < MTILE) sS[t] = 0.0f;

    // ---- stage A tile: h_enc[b, l0:l0+128, :] fp32 -> bf16 in LDS ----
    {
        const float4* src = (const float4*)(h_enc + ((size_t)b * L_DIM + l0) * D_DIM);
        uint2* dst = (uint2*)sA;
        for (int i = t; i < (MTILE * D_DIM) / 4; i += 256) {
            float4 f = src[i];
            union { __bf16 h[4]; uint2 u; } cv;
            cv.h[0] = (__bf16)f.x; cv.h[1] = (__bf16)f.y;
            cv.h[2] = (__bf16)f.z; cv.h[3] = (__bf16)f.w;
            dst[i] = cv.u;
        }
    }

    union ABFrag { v16bf v; uint4 q[2]; };
    const char* w1b  = (const char*)w1bf;
    const char* aRow = sA + (size_t)(16 * wv + n) * (D_DIM * 2);
    const u32x8 g1   = tdm_group1();

    float sacc[8];
#pragma unroll
    for (int r = 0; r < 8; ++r) sacc[r] = 0.0f;

    auto score_tile = [&](v8f a, int col) {
        float dv = sDPB[col], vw = sV[col];
#pragma unroll
        for (int r = 0; r < 8; ++r) {
            float x = a[r] + dv;               // enc_proj + b1 + dec_proj
            sacc[r] += fast_tanh(x) * vw;      // row = 16wv + 8g + r
        }
    };

#pragma unroll 1
    for (int nc = 0; nc < D_DIM / NCHUNK; ++nc) {
        const int n0 = nc * NCHUNK;
        v8f acc0 = {}, acc1 = {}, acc2 = {}, acc3 = {};

        if (wv == 0)   // prologue: block 0 -> buf 0 (one DMA for the whole tile)
            tdm_load_b_block(w1b + (size_t)n0 * (D_DIM * 2), ldsB, g1);

#pragma unroll 1
        for (int kb = 0; kb < D_DIM / KBLOCK; ++kb) {
            if (wv == 0) __builtin_amdgcn_s_wait_tensorcnt(0);  // block kb landed
            __syncthreads();      // publish to all waves; prev buffer fully consumed
            if (wv == 0 && kb + 1 < D_DIM / KBLOCK)
                tdm_load_b_block(w1b + (size_t)n0 * (D_DIM * 2) + (size_t)(kb + 1) * KBLOCK * 2,
                                 ldsB + ((kb + 1) & 1) * SMEM_BBUF, g1);

            const char* bufp = sB + (kb & 1) * SMEM_BBUF;
#pragma unroll
            for (int ks = 0; ks < KBLOCK; ks += 32) {
                const int kloc = kb * KBLOCK + ks;
                // A fragment (16x32 bf16): halves 0..7 hold K=kloc+16g+{0..7},
                // halves 8..15 hold K=kloc+16g+8+{0..7}.
                ABFrag af;
                af.q[0] = *(const uint4*)(aRow + (kloc + 16 * g) * 2);
                af.q[1] = *(const uint4*)(aRow + (kloc + 16 * g + 8) * 2);
                // B fragments: lane(n,g) holds column n of tile, K = ks+16g+h.
                const char* bbase = bufp + (ks + 16 * g) * 2;
                ABFrag bf0, bf1, bf2, bf3;
                bf0.q[0] = *(const uint4*)(bbase + (0 * 16 + n) * 256);
                bf0.q[1] = *(const uint4*)(bbase + (0 * 16 + n) * 256 + 16);
                bf1.q[0] = *(const uint4*)(bbase + (1 * 16 + n) * 256);
                bf1.q[1] = *(const uint4*)(bbase + (1 * 16 + n) * 256 + 16);
                bf2.q[0] = *(const uint4*)(bbase + (2 * 16 + n) * 256);
                bf2.q[1] = *(const uint4*)(bbase + (2 * 16 + n) * 256 + 16);
                bf3.q[0] = *(const uint4*)(bbase + (3 * 16 + n) * 256);
                bf3.q[1] = *(const uint4*)(bbase + (3 * 16 + n) * 256 + 16);
                acc0 = __builtin_amdgcn_wmma_f32_16x16x32_bf16(false, af.v, false, bf0.v,
                                                               (short)0, acc0, false, false);
                acc1 = __builtin_amdgcn_wmma_f32_16x16x32_bf16(false, af.v, false, bf1.v,
                                                               (short)0, acc1, false, false);
                acc2 = __builtin_amdgcn_wmma_f32_16x16x32_bf16(false, af.v, false, bf2.v,
                                                               (short)0, acc2, false, false);
                acc3 = __builtin_amdgcn_wmma_f32_16x16x32_bf16(false, af.v, false, bf3.v,
                                                               (short)0, acc3, false, false);
            }
        }
        // ---- epilogue for this 64-wide chunk: tanh + dot with V ----
        score_tile(acc0, n0 + 0 * 16 + n);
        score_tile(acc1, n0 + 1 * 16 + n);
        score_tile(acc2, n0 + 2 * 16 + n);
        score_tile(acc3, n0 + 3 * 16 + n);
    }

    // ---- reduce per-row partial scores across lanes via LDS ----
#pragma unroll
    for (int r = 0; r < 8; ++r)
        atomicAdd(&sS[16 * wv + 8 * g + r], sacc[r]);
    __syncthreads();
    if (t < MTILE) scores[(size_t)b * L_DIM + l0 + t] = sS[t];
}

// ---------------- K4: softmax over L per batch ---------------------------
__global__ __launch_bounds__(256) void softmax_kernel(const float* __restrict__ scores,
                                                      float* __restrict__ attn) {
    __shared__ float red[256];
    int b = blockIdx.x, t = threadIdx.x;
    const float* s = scores + (size_t)b * L_DIM;
    float m = -1e30f;
    for (int i = t; i < L_DIM; i += 256) m = fmaxf(m, s[i]);
    red[t] = m; __syncthreads();
    for (int o = 128; o > 0; o >>= 1) {
        if (t < o) red[t] = fmaxf(red[t], red[t + o]);
        __syncthreads();
    }
    m = red[0]; __syncthreads();
    float sum = 0.0f;
    for (int i = t; i < L_DIM; i += 256) {
        float e = __expf(s[i] - m);
        attn[(size_t)b * L_DIM + i] = e;
        sum += e;
    }
    red[t] = sum; __syncthreads();
    for (int o = 128; o > 0; o >>= 1) {
        if (t < o) red[t] += red[t + o];
        __syncthreads();
    }
    float inv = 1.0f / red[0];
    for (int i = t; i < L_DIM; i += 256) attn[(size_t)b * L_DIM + i] *= inv;
}

// ---------------- K5: gT[d][b] = sum_l attn[b][l] * h_enc[b][l][d] -------
__global__ __launch_bounds__(256) void wsum_kernel(const float* __restrict__ h_enc,
                                                   const float* __restrict__ attn,
                                                   float* __restrict__ gT) {
    int b = blockIdx.y;
    int d = blockIdx.x * 256 + threadIdx.x;
    const float* he = h_enc + (size_t)b * L_DIM * D_DIM + d;
    const float* a  = attn + (size_t)b * L_DIM;
    float acc = 0.0f;
#pragma unroll 4
    for (int l = 0; l < L_DIM; ++l) acc += a[l] * he[(size_t)l * D_DIM];
    gT[(size_t)d * B_DIM + b] = acc;
}

// ---------------- K6: ctx[b][e] = sum_d W1[e][d]*gT[d][b] + b1[e] --------
__global__ __launch_bounds__(256) void ctx_kernel(const float* __restrict__ W1,
                                                  const float* __restrict__ b1,
                                                  const float* __restrict__ gT,
                                                  float* __restrict__ out) {
    int idx = blockIdx.x * 256 + threadIdx.x;    // 32768 outputs
    int b = idx & 31, e = idx >> 5;
    const float* wr = W1 + (size_t)e * D_DIM;
    float acc = 0.0f;
#pragma unroll 4
    for (int d = 0; d < D_DIM; ++d) acc += wr[d] * gT[(size_t)d * B_DIM + b];
    out[(size_t)b * D_DIM + e] = acc + b1[e];
}

// ---------------- host launcher ------------------------------------------
extern "C" void kernel_launch(void* const* d_in, const int* in_sizes, int n_in,
                              void* d_out, int out_size, void* d_ws, size_t ws_size,
                              hipStream_t stream) {
    const float* h_enc = (const float*)d_in[0];
    const float* h_dec = (const float*)d_in[1];
    const float* W1    = (const float*)d_in[2];
    const float* b1    = (const float*)d_in[3];
    const float* W2    = (const float*)d_in[4];
    const float* b2    = (const float*)d_in[5];
    const float* Vv    = (const float*)d_in[6];
    // bv (d_in[7]) is a constant added to all scores -> cancels in softmax.

    char* ws = (char*)d_ws;
    __bf16* w1bf  = (__bf16*)ws;                                   // 2 MB
    float*  dpb   = (float*)(ws + (2u << 20));                     // 128 KB
    float*  score = (float*)(ws + (2u << 20) + (128u << 10));      // 256 KB
    float*  attn  = score + B_DIM * L_DIM;                         // 256 KB
    float*  gT    = attn + B_DIM * L_DIM;                          // 128 KB

    (void)hipFuncSetAttribute((const void*)attn_scores_kernel,
                              hipFuncAttributeMaxDynamicSharedMemorySize, SMEM_BYTES);

    cvt_w1_kernel<<<1024, 256, 0, stream>>>(W1, w1bf);
    decproj_kernel<<<128, 256, 0, stream>>>(h_dec, W2, b1, b2, dpb);
    attn_scores_kernel<<<dim3(L_DIM / MTILE, B_DIM), 256, SMEM_BYTES, stream>>>(
        h_enc, w1bf, dpb, Vv, score);
    softmax_kernel<<<B_DIM, 256, 0, stream>>>(score, attn);
    wsum_kernel<<<dim3(D_DIM / 256, B_DIM), 256, 0, stream>>>(h_enc, attn, gT);
    ctx_kernel<<<128, 256, 0, stream>>>(W1, b1, gT, (float*)d_out);
}
